// MinMaxRayAttn_Net_37263136260938
// MI455X (gfx1250) — matrix operand order repaired and checked by reading, not compiled, not run on hardware
//
#include <hip/hip_runtime.h>
#include <hip/hip_bf16.h>

// ---------------------------------------------------------------------------
// Types for CDNA5 WMMA (gfx1250, wave32)
// ---------------------------------------------------------------------------
typedef __attribute__((ext_vector_type(16))) __bf16 v16bf;
typedef __attribute__((ext_vector_type(8)))  __bf16 bf16x8;
typedef __attribute__((ext_vector_type(4)))  __bf16 bf16x4;
typedef __attribute__((ext_vector_type(8)))  float  v8f;
typedef __attribute__((ext_vector_type(4)))  float  f32x4;

__device__ __forceinline__ __bf16 f2bf(float x) { return (__bf16)x; }
__device__ __forceinline__ bf16x4 cvt4(f32x4 v) { return __builtin_convertvector(v, bf16x4); }
__device__ __forceinline__ v8f    up8(bf16x8 v) { return __builtin_convertvector(v, v8f); }
__device__ __forceinline__ bf16x8 dn8(v8f v)    { return __builtin_convertvector(v, bf16x8); }

// Load a 16-element bf16 fragment as two contiguous 16B chunks from LDS.
__device__ __forceinline__ v16bf ldfrag(const __bf16* base, int k0, int k1) {
  union { bf16x8 h[2]; v16bf v; } u;
  u.h[0] = *(const bf16x8*)(base + k0);
  u.h[1] = *(const bf16x8*)(base + k1);
  return u.v;
}

#define WMMA_BF16(a, b, c) \
  __builtin_amdgcn_wmma_f32_16x16x32_bf16(false, (a), false, (b), (short)0, (c), false, false)

// ---------------------------------------------------------------------------
// FAST GEMM (bf16-resident): A bf16 [M,K], W f32 [K,N], C bf16 [M,N].
// Requires K%32==0, M%128==0. Tile 128x64, BK=32, 8 waves, 4 WMMAs/K-step.
// A staging is a pure b128 copy; W staged with packed f32->bf16 conversion.
// Double-buffered LDS + register prefetch + L2 prefetch of tile after next.
// ---------------------------------------------------------------------------
__global__ __launch_bounds__(256) void gemm_fast_kernel(
    const __bf16* __restrict__ A, const float* __restrict__ Wt,
    const float* __restrict__ bias, const __bf16* __restrict__ res,
    __bf16* __restrict__ C, int M, int K, int N, int do_relu)
{
  __shared__ __attribute__((aligned(16))) __bf16 As[2][128 * 32];  // [m][k]
  __shared__ __attribute__((aligned(16))) __bf16 Bs[2][64 * 32];   // [n][k]

  const int t    = threadIdx.x;
  const int mRow = blockIdx.y * 128;
  const int nCol = blockIdx.x * 64;
  const int w    = t >> 5;
  const int lane = t & 31;
  const int hl   = lane >> 4;
  const int l16  = lane & 15;
  const int mW   = (w & 3) * 32;
  const int nW   = (w >> 2) * 32;

  bf16x8 aReg[2];
  f32x4  bReg[2];

  auto loadA = [&](int k0) {
#pragma unroll
    for (int e = 0; e < 2; ++e) {
      int g = e * 256 + t;             // 8-elem group over 128x4
      int r = g >> 2, c8 = (g & 3) * 8;
      aReg[e] = *(const bf16x8*)(A + (size_t)(mRow + r) * K + k0 + c8);
    }
  };
  auto loadB = [&](int k0) {
#pragma unroll
    for (int e = 0; e < 2; ++e) {
      int idx = e * 256 + t;           // float4 index over 32x16
      int r = idx >> 4, c4 = idx & 15;
      int gn = nCol + c4 * 4;
      f32x4 z = {0.f, 0.f, 0.f, 0.f};
      bReg[e] = (gn + 3 < N) ? *(const f32x4*)(Wt + (size_t)(k0 + r) * N + gn) : z;
    }
  };
  auto storeA = [&](int buf) {
#pragma unroll
    for (int e = 0; e < 2; ++e) {
      int g = e * 256 + t;
      int r = g >> 2, c8 = (g & 3) * 8;
      *(bf16x8*)&As[buf][r * 32 + c8] = aReg[e];           // b128 copy
    }
  };
  auto storeB = [&](int buf) {
#pragma unroll
    for (int e = 0; e < 2; ++e) {
      int idx = e * 256 + t;
      int r = idx >> 4, c4 = idx & 15;
      bf16x4 h = cvt4(bReg[e]);
#pragma unroll
      for (int j = 0; j < 4; ++j)     // transpose scatter [k][n] -> [n][k]
        Bs[buf][(c4 * 4 + j) * 32 + r] = h[j];
    }
  };

  const int nK = K >> 5;
  loadA(0); loadB(0);
  storeA(0); storeB(0);
  __syncthreads();

  v8f acc[4] = {{}, {}, {}, {}};
  for (int kb = 0; kb < nK; ++kb) {
    const int cur = kb & 1, nxt = cur ^ 1;
    const bool more = (kb + 1) < nK;
    if (more) { loadA((kb + 1) * 32); loadB((kb + 1) * 32); }
    if (kb + 2 < nK) {
      int k2 = (kb + 2) * 32;
      __builtin_prefetch(A + (size_t)(mRow + (t >> 2)) * K + k2 + (t & 3) * 8, 0, 1);
      __builtin_prefetch(Wt + (size_t)(k2 + (t >> 4)) * N + nCol + (t & 15) * 4, 0, 1);
    }

    const __bf16* Ab = As[cur];
    const __bf16* Bb = Bs[cur];
    v16bf a0 = ldfrag(&Ab[(mW + l16) * 32],      hl * 16, hl * 16 + 8);
    v16bf a1 = ldfrag(&Ab[(mW + 16 + l16) * 32], hl * 16, hl * 16 + 8);
    v16bf b0 = ldfrag(&Bb[(nW + l16) * 32],      hl * 16, hl * 16 + 8);
    v16bf b1 = ldfrag(&Bb[(nW + 16 + l16) * 32], hl * 16, hl * 16 + 8);

    acc[0] = WMMA_BF16(a0, b0, acc[0]);
    acc[1] = WMMA_BF16(a0, b1, acc[1]);
    acc[2] = WMMA_BF16(a1, b0, acc[2]);
    acc[3] = WMMA_BF16(a1, b1, acc[3]);

    if (more) { storeA(nxt); storeB(nxt); }
    __syncthreads();
  }

#pragma unroll
  for (int mi = 0; mi < 2; ++mi) {
#pragma unroll
    for (int nb = 0; nb < 2; ++nb) {
      int gn = nCol + nW + nb * 16 + l16;
      if (gn >= N) continue;
      float bv = bias ? bias[gn] : 0.f;
      v8f a = acc[mi * 2 + nb];
#pragma unroll
      for (int r = 0; r < 8; ++r) {
        int gm = mRow + mW + mi * 16 + hl * 8 + r;
        float v = a[r] + bv;
        if (do_relu) v = fmaxf(v, 0.f);
        size_t ci = (size_t)gm * N + gn;
        if (res) v += (float)res[ci];
        C[ci] = f2bf(v);
      }
    }
  }
}

// ---------------------------------------------------------------------------
// Generic GEMM (ragged K: 35, 4, 346; fp32 A). Tile 64x64, bf16 C out.
// ---------------------------------------------------------------------------
__global__ __launch_bounds__(256) void gemm_kernel(
    const float* __restrict__ A, const float* __restrict__ Wt,
    const float* __restrict__ bias, const __bf16* __restrict__ res,
    __bf16* __restrict__ C, int M, int K, int N, int do_relu)
{
  __shared__ __attribute__((aligned(16))) __bf16 As[64 * 32];
  __shared__ __attribute__((aligned(16))) __bf16 Bs[64 * 32];

  const int t    = threadIdx.x;
  const int mRow = blockIdx.y * 64;
  const int nCol = blockIdx.x * 64;
  const int w    = t >> 5;
  const int lane = t & 31;
  const int hl   = lane >> 4;
  const int l16  = lane & 15;
  const int mW   = (w & 3) * 16;
  const int nW   = (w >> 2) * 32;

  v8f acc0 = {}, acc1 = {};
  const int nK = (K + 31) >> 5;

  for (int kb = 0; kb < nK; ++kb) {
    const int k0 = kb * 32;
#pragma unroll
    for (int e = 0; e < 8; ++e) {
      int i = e * 256 + t;
      int r = i >> 5, c = i & 31;
      int gk = k0 + c;
      float v = (gk < K) ? A[(size_t)(mRow + r) * K + gk] : 0.f;
      As[r * 32 + c] = f2bf(v);
    }
#pragma unroll
    for (int e = 0; e < 8; ++e) {
      int i = e * 256 + t;
      int r = i >> 6, c = i & 63;
      int gk = k0 + r, gn = nCol + c;
      float v = (gk < K && gn < N) ? Wt[(size_t)gk * N + gn] : 0.f;
      Bs[c * 32 + r] = f2bf(v);
    }
    __syncthreads();

    v16bf a  = ldfrag(&As[(mW + l16) * 32], hl * 16, hl * 16 + 8);
    v16bf b0 = ldfrag(&Bs[(nW + l16) * 32], hl * 16, hl * 16 + 8);
    v16bf b1 = ldfrag(&Bs[(nW + 16 + l16) * 32], hl * 16, hl * 16 + 8);

    acc0 = WMMA_BF16(a, b0, acc0);
    acc1 = WMMA_BF16(a, b1, acc1);
    __syncthreads();
  }

#pragma unroll
  for (int nb = 0; nb < 2; ++nb) {
    int gn = nCol + nW + nb * 16 + l16;
    if (gn >= N) continue;
    float bv = bias ? bias[gn] : 0.f;
    v8f acc = nb ? acc1 : acc0;
#pragma unroll
    for (int r = 0; r < 8; ++r) {
      int gm = mRow + mW + hl * 8 + r;
      float v = acc[r] + bv;
      if (do_relu) v = fmaxf(v, 0.f);
      size_t ci = (size_t)gm * N + gn;
      if (res) v += (float)res[ci];
      C[ci] = f2bf(v);
    }
  }
}

// ---------------------------------------------------------------------------
// Self attention: one block per (b,h). bf16 in/out, zero-conversion staging.
// ---------------------------------------------------------------------------
__global__ __launch_bounds__(256) void self_attn_kernel(
    const __bf16* __restrict__ qg, const __bf16* __restrict__ kg,
    const __bf16* __restrict__ vg, __bf16* __restrict__ og)
{
  __shared__ __attribute__((aligned(16))) __bf16 Qs[64 * 64];  // [s][d]
  __shared__ __attribute__((aligned(16))) __bf16 Ks[64 * 64];  // [s][d]
  __shared__ __attribute__((aligned(16))) __bf16 Vt[64 * 64];  // [d][s]
  __shared__ float Pf[64 * 64];
  __shared__ __attribute__((aligned(16))) __bf16 Ps[64 * 64];

  const int b  = blockIdx.x >> 2;
  const int h  = blockIdx.x & 3;
  const int co = h * 64;
  const int t  = threadIdx.x;

#pragma unroll
  for (int e = 0; e < 2; ++e) {
    int g = e * 256 + t;               // 8-elem groups over 64x8
    int s = g >> 3, d8 = (g & 7) * 8;
    size_t gi = (size_t)(b * 64 + s) * 256 + co + d8;
    bf16x8 q8 = *(const bf16x8*)(qg + gi);
    bf16x8 k8 = *(const bf16x8*)(kg + gi);
    bf16x8 v8 = *(const bf16x8*)(vg + gi);
    *(bf16x8*)&Qs[s * 64 + d8] = q8;
    *(bf16x8*)&Ks[s * 64 + d8] = k8;
#pragma unroll
    for (int j = 0; j < 8; ++j) Vt[(d8 + j) * 64 + s] = v8[j];
  }
  __syncthreads();

  const int w = t >> 5, lane = t & 31, hl = lane >> 4, l16 = lane & 15;
  const int mW = (w & 3) * 16;
  const int nW = (w >> 2) * 32;

  v8f s0 = {}, s1 = {};
#pragma unroll
  for (int kk = 0; kk < 64; kk += 32) {
    v16bf a  = ldfrag(&Qs[(mW + l16) * 64], kk + hl * 16, kk + hl * 16 + 8);
    v16bf b0 = ldfrag(&Ks[(nW + l16) * 64], kk + hl * 16, kk + hl * 16 + 8);
    v16bf b1 = ldfrag(&Ks[(nW + 16 + l16) * 64], kk + hl * 16, kk + hl * 16 + 8);
    s0 = WMMA_BF16(a, b0, s0);
    s1 = WMMA_BF16(a, b1, s1);
  }
#pragma unroll
  for (int r = 0; r < 8; ++r) {
    int m = mW + hl * 8 + r;
    Pf[m * 64 + nW + l16]      = s0[r] * 0.125f;  // 1/sqrt(64)
    Pf[m * 64 + nW + 16 + l16] = s1[r] * 0.125f;
  }
  __syncthreads();

  if (t < 64) {
    float mx = -3.4e38f;
    for (int j = 0; j < 64; ++j) mx = fmaxf(mx, Pf[t * 64 + j]);
    float ssum = 0.f;
    for (int j = 0; j < 64; ++j) ssum += expf(Pf[t * 64 + j] - mx);
    float inv = 1.f / ssum;
    for (int j = 0; j < 64; ++j)
      Ps[t * 64 + j] = f2bf(expf(Pf[t * 64 + j] - mx) * inv);
  }
  __syncthreads();

  v8f o0 = {}, o1 = {};
#pragma unroll
  for (int kk = 0; kk < 64; kk += 32) {
    v16bf a  = ldfrag(&Ps[(mW + l16) * 64], kk + hl * 16, kk + hl * 16 + 8);
    v16bf b0 = ldfrag(&Vt[(nW + l16) * 64], kk + hl * 16, kk + hl * 16 + 8);
    v16bf b1 = ldfrag(&Vt[(nW + 16 + l16) * 64], kk + hl * 16, kk + hl * 16 + 8);
    o0 = WMMA_BF16(a, b0, o0);
    o1 = WMMA_BF16(a, b1, o1);
  }
#pragma unroll
  for (int r = 0; r < 8; ++r) {
    int m = mW + hl * 8 + r;
    og[(size_t)(b * 64 + m) * 256 + co + nW + l16]      = f2bf(o0[r]);
    og[(size_t)(b * 64 + m) * 256 + co + nW + 16 + l16] = f2bf(o1[r]);
  }
}

// ---------------------------------------------------------------------------
// LayerNorm over 256 channels: one wave32 per row, bf16x8 I/O, fp32 math.
// ---------------------------------------------------------------------------
__global__ __launch_bounds__(256) void ln_kernel(
    const __bf16* __restrict__ x, const float* __restrict__ g,
    const float* __restrict__ b, __bf16* __restrict__ y, int rows, float eps)
{
  int w = threadIdx.x >> 5, lane = threadIdx.x & 31;
  int row = blockIdx.x * 8 + w;
  if (row >= rows) return;
  const __bf16* xr = x + (size_t)row * 256;
  v8f v = up8(*(const bf16x8*)(xr + lane * 8));
  float s = 0.f;
#pragma unroll
  for (int j = 0; j < 8; ++j) s += v[j];
#pragma unroll
  for (int off = 16; off; off >>= 1) s += __shfl_xor(s, off, 32);
  float m = s * (1.f / 256.f);
  float q = 0.f;
#pragma unroll
  for (int j = 0; j < 8; ++j) { float d = v[j] - m; q += d * d; }
#pragma unroll
  for (int off = 16; off; off >>= 1) q += __shfl_xor(q, off, 32);
  float r = rsqrtf(q * (1.f / 256.f) + eps);
  f32x4 g0 = *(const f32x4*)(g + lane * 8);
  f32x4 g1 = *(const f32x4*)(g + lane * 8 + 4);
  f32x4 b0 = *(const f32x4*)(b + lane * 8);
  f32x4 b1 = *(const f32x4*)(b + lane * 8 + 4);
  v8f o;
#pragma unroll
  for (int j = 0; j < 4; ++j) {
    o[j]     = (v[j]     - m) * r * g0[j] + b0[j];
    o[j + 4] = (v[j + 4] - m) * r * g1[j] + b1[j];
  }
  *(bf16x8*)(y + (size_t)row * 256 + lane * 8) = dn8(o);
}

// fq[bs, w8] = max over V (8-wide bf16)
__global__ void maxpool_kernel(const __bf16* __restrict__ rgbf, __bf16* __restrict__ fq) {
  int i8 = blockIdx.x * blockDim.x + threadIdx.x;
  if (i8 >= 4096 * 32) return;
  int bs = i8 >> 5, w8 = (i8 & 31) * 8;
  v8f m = up8(*(const bf16x8*)(rgbf + (size_t)(bs << 3) * 256 + w8));
#pragma unroll
  for (int v = 1; v < 8; ++v) {
    v8f x = up8(*(const bf16x8*)(rgbf + (size_t)((bs << 3) + v) * 256 + w8));
#pragma unroll
    for (int j = 0; j < 8; ++j) m[j] = fmaxf(m[j], x[j]);
  }
  *(bf16x8*)(fq + (size_t)bs * 256 + w8) = dn8(m);
}

// t = k - q(broadcast over views) + pos (8-wide bf16)
__global__ void tdiff_kernel(const __bf16* __restrict__ k, const __bf16* __restrict__ q,
                             const __bf16* __restrict__ pos, __bf16* __restrict__ t_) {
  int i8 = blockIdx.x * blockDim.x + threadIdx.x;
  if (i8 >= 32768 * 32) return;
  int w8 = (i8 & 31) * 8;
  int bsv = i8 >> 5, bs = bsv >> 3;
  size_t gi = (size_t)bsv * 256 + w8;
  v8f kv = up8(*(const bf16x8*)(k + gi));
  v8f qv = up8(*(const bf16x8*)(q + (size_t)bs * 256 + w8));
  v8f pv = up8(*(const bf16x8*)(pos + gi));
  *(bf16x8*)(t_ + gi) = dn8(kv - qv + pv);
}

// per-channel softmax over views with mask, weighted sum of (v + pos); 8-wide
__global__ void cross_combine_kernel(const __bf16* __restrict__ a, const __bf16* __restrict__ vv,
                                     const __bf16* __restrict__ pos, const int* __restrict__ mask,
                                     __bf16* __restrict__ o) {
  int i8 = blockIdx.x * blockDim.x + threadIdx.x;
  if (i8 >= 4096 * 32) return;
  int bs = i8 >> 5, w8 = (i8 & 31) * 8;
  v8f lg[8];
  v8f mx = {-3.4e38f, -3.4e38f, -3.4e38f, -3.4e38f, -3.4e38f, -3.4e38f, -3.4e38f, -3.4e38f};
  v8f neg = {-1e9f, -1e9f, -1e9f, -1e9f, -1e9f, -1e9f, -1e9f, -1e9f};
#pragma unroll
  for (int v = 0; v < 8; ++v) {
    int mi = (bs << 3) + v;
    v8f l = (mask[mi] == 0) ? neg : up8(*(const bf16x8*)(a + (size_t)mi * 256 + w8));
    lg[v] = l;
#pragma unroll
    for (int j = 0; j < 8; ++j) mx[j] = fmaxf(mx[j], l[j]);
  }
  v8f s = {};
#pragma unroll
  for (int v = 0; v < 8; ++v) {
#pragma unroll
    for (int j = 0; j < 8; ++j) { lg[v][j] = expf(lg[v][j] - mx[j]); s[j] += lg[v][j]; }
  }
  v8f acc = {};
#pragma unroll
  for (int v = 0; v < 8; ++v) {
    size_t ai = (size_t)((bs << 3) + v) * 256 + w8;
    v8f vp = up8(*(const bf16x8*)(vv + ai));
    v8f pp = up8(*(const bf16x8*)(pos + ai));
#pragma unroll
    for (int j = 0; j < 8; ++j) acc[j] += (vp[j] + pp[j]) * lg[v][j] / s[j];
  }
  *(bf16x8*)(o + (size_t)bs * 256 + w8) = dn8(acc);
}

__global__ void penc_pts_kernel(const float* __restrict__ pts, float* __restrict__ out) {
  int bs = blockIdx.x * blockDim.x + threadIdx.x;
  if (bs >= 4096) return;
  float x[3];
#pragma unroll
  for (int d = 0; d < 3; ++d) x[d] = pts[bs * 3 + d];
  float* o = out + (size_t)bs * 63;
#pragma unroll
  for (int d = 0; d < 3; ++d) o[d] = x[d];
  for (int d = 0; d < 3; ++d) {
    float f = 1.f;
    for (int k = 0; k < 10; ++k) {
      float a = x[d] * f;
      o[3 + d * 10 + k]  = sinf(a);
      o[33 + d * 10 + k] = cosf(a);
      f *= 2.f;
    }
  }
}

__global__ void penc_views_kernel(const float* __restrict__ ray_d, float* __restrict__ out) {
  int b = threadIdx.x;
  if (b >= 64) return;
  float x0 = ray_d[b * 3], x1 = ray_d[b * 3 + 1], x2 = ray_d[b * 3 + 2];
  float inv = rsqrtf(x0 * x0 + x1 * x1 + x2 * x2);
  float v[3] = {x0 * inv, x1 * inv, x2 * inv};
  float* o = out + (size_t)b * 27;
#pragma unroll
  for (int d = 0; d < 3; ++d) o[d] = v[d];
  for (int d = 0; d < 3; ++d) {
    float f = 1.f;
    for (int k = 0; k < 4; ++k) {
      float a = v[d] * f;
      o[3 + d * 4 + k]  = sinf(a);
      o[15 + d * 4 + k] = cosf(a);
      f *= 2.f;
    }
  }
}

// cat = [fq(256, bf16) | pts_enc(63, f32) | views_enc(27, f32)] -> f32 [4096,346]
__global__ void concat_kernel(const __bf16* __restrict__ fq, const float* __restrict__ pe,
                              const float* __restrict__ ve, float* __restrict__ cat) {
  int idx = blockIdx.x * blockDim.x + threadIdx.x;
  if (idx >= 4096 * 346) return;
  int row = idx / 346, c = idx % 346;
  float v;
  if (c < 256)      v = (float)fq[(size_t)row * 256 + c];
  else if (c < 319) v = pe[(size_t)row * 63 + (c - 256)];
  else              v = ve[(size_t)(row >> 6) * 27 + (c - 319)];
  cat[idx] = v;
}

__global__ __launch_bounds__(256) void final_kernel(
    const __bf16* __restrict__ fq, const float* __restrict__ g, const float* __restrict__ b,
    const float* __restrict__ ow, const float* __restrict__ ob, float* __restrict__ out)
{
  __shared__ float red[256];
  __shared__ float pool[256];
  int bb = blockIdx.x, c = threadIdx.x;
  float acc = 0.f;
  for (int s = 0; s < 64; ++s) {
    float x = (float)fq[(size_t)(bb * 64 + s) * 256 + c];
    red[c] = x; __syncthreads();
    for (int off = 128; off > 0; off >>= 1) { if (c < off) red[c] += red[c + off]; __syncthreads(); }
    float m = red[0] * (1.f / 256.f); __syncthreads();
    float d = x - m;
    red[c] = d * d; __syncthreads();
    for (int off = 128; off > 0; off >>= 1) { if (c < off) red[c] += red[c + off]; __syncthreads(); }
    float v = red[0] * (1.f / 256.f); __syncthreads();
    acc += d * rsqrtf(v + 1e-5f) * g[c] + b[c];
  }
  pool[c] = acc * (1.f / 64.f);
  __syncthreads();
  if (c < 3) {
    float o = ob[c];
    for (int k = 0; k < 256; ++k) o += pool[k] * ow[k * 3 + c];
    out[bb * 3 + c] = o;
  }
}

// ---------------------------------------------------------------------------
// Host orchestration
// ---------------------------------------------------------------------------
extern "C" void kernel_launch(void* const* d_in, const int* in_sizes, int n_in,
                              void* d_out, int out_size, void* d_ws, size_t ws_size,
                              hipStream_t stream) {
  (void)in_sizes; (void)n_in; (void)out_size; (void)ws_size;
  const float* rgb_feat = (const float*)d_in[0];
  const float* ray_diff = (const float*)d_in[1];
  const int*   mask     = (const int*)d_in[2];
  const float* pts      = (const float*)d_in[3];
  const float* ray_d    = (const float*)d_in[4];
  const float* rf_w1 = (const float*)d_in[5];
  const float* rf_b1 = (const float*)d_in[6];
  const float* rf_w2 = (const float*)d_in[7];
  const float* rf_b2 = (const float*)d_in[8];
  const float* c_ln_g = (const float*)d_in[9];
  const float* c_ln_b = (const float*)d_in[10];
  const float* c_qw = (const float*)d_in[11];
  const float* c_kw = (const float*)d_in[12];
  const float* c_vw = (const float*)d_in[13];
  const float* c_pw1 = (const float*)d_in[14];
  const float* c_pb1 = (const float*)d_in[15];
  const float* c_pw2 = (const float*)d_in[16];
  const float* c_pb2 = (const float*)d_in[17];
  const float* c_aw1 = (const float*)d_in[18];
  const float* c_ab1 = (const float*)d_in[19];
  const float* c_aw2 = (const float*)d_in[20];
  const float* c_ab2 = (const float*)d_in[21];
  const float* c_ow = (const float*)d_in[22];
  const float* c_ob = (const float*)d_in[23];
  const float* c_fg = (const float*)d_in[24];
  const float* c_fb = (const float*)d_in[25];
  const float* c_fw1 = (const float*)d_in[26];
  const float* c_fb1 = (const float*)d_in[27];
  const float* c_fw2 = (const float*)d_in[28];
  const float* c_fb2 = (const float*)d_in[29];
  const float* s_ln_g = (const float*)d_in[30];
  const float* s_ln_b = (const float*)d_in[31];
  const float* s_qw = (const float*)d_in[32];
  const float* s_kw = (const float*)d_in[33];
  const float* s_vw = (const float*)d_in[34];
  const float* s_ow = (const float*)d_in[35];
  const float* s_ob = (const float*)d_in[36];
  const float* s_fg = (const float*)d_in[37];
  const float* s_fb = (const float*)d_in[38];
  const float* s_fw1 = (const float*)d_in[39];
  const float* s_fb1 = (const float*)d_in[40];
  const float* s_fw2 = (const float*)d_in[41];
  const float* s_fb2 = (const float*)d_in[42];
  const float* q_w1 = (const float*)d_in[43];
  const float* q_b1 = (const float*)d_in[44];
  const float* q_w2 = (const float*)d_in[45];
  const float* q_b2 = (const float*)d_in[46];
  const float* norm_g = (const float*)d_in[47];
  const float* norm_b = (const float*)d_in[48];
  const float* out_w = (const float*)d_in[49];
  const float* out_b = (const float*)d_in[50];

  // Workspace carve-up. Activations are bf16; penc/concat staging is f32.
  const size_t BSVW = 32768ull * 256;
  const size_t BSW  = 4096ull * 256;
  __bf16* wsb  = (__bf16*)d_ws;
  __bf16* rgbf = wsb;
  __bf16* bufK = rgbf + BSVW;
  __bf16* bufV = bufK + BSVW;
  __bf16* bufP = bufV + BSVW;
  __bf16* bufT = bufP + BSVW;
  __bf16* ahid = bufT + BSVW;              // [32768,32]
  __bf16* fq   = ahid + 32768ull * 32;
  __bf16* xq   = fq + BSW;
  __bf16* qb   = xq + BSW;
  __bf16* ob   = qb + BSW;
  __bf16* hid  = ob + BSW;                 // [4096,1024]
  __bf16* sq   = hid + 4096ull * 1024;
  __bf16* sk   = sq + BSW;
  __bf16* sv   = sk + BSW;
  float*  catb = (float*)(sv + BSW);       // f32 [4096,346]
  float*  pe   = catb + 4096ull * 346;     // f32 [4096,63]
  float*  ve   = pe + 4096ull * 63;        // f32 [64,27]

  auto gemmF = [&](const __bf16* A, const float* Wt, const float* bias, const __bf16* res,
                   __bf16* C, int M, int K, int N, int relu) {
    dim3 grid((unsigned)((N + 63) / 64), (unsigned)(M / 128));
    gemm_fast_kernel<<<grid, 256, 0, stream>>>(A, Wt, bias, res, C, M, K, N, relu);
  };
  auto gemmG = [&](const float* A, const float* Wt, const float* bias, const __bf16* res,
                   __bf16* C, int M, int K, int N, int relu) {
    dim3 grid((unsigned)((N + 63) / 64), (unsigned)((M + 63) / 64));
    gemm_kernel<<<grid, 256, 0, stream>>>(A, Wt, bias, res, C, M, K, N, relu);
  };
  auto ln = [&](const __bf16* x, const float* g, const float* b, __bf16* y, float eps) {
    ln_kernel<<<512, 256, 0, stream>>>(x, g, b, y, 4096, eps);
  };

  // ---- prologue ----
  penc_pts_kernel<<<16, 256, 0, stream>>>(pts, pe);
  penc_views_kernel<<<1, 64, 0, stream>>>(ray_d, ve);
  gemmG(rgb_feat, rf_w1, rf_b1, nullptr, bufK, 32768, 35, 256, 1);
  gemmF(bufK, rf_w2, rf_b2, nullptr, rgbf, 32768, 256, 256, 0);
  maxpool_kernel<<<512, 256, 0, stream>>>(rgbf, fq);

  int j = 0;
  for (int i = 0; i < 8; ++i) {
    // ---- cross Transformer2D (ray attention over views) ----
    ln(fq, c_ln_g + i * 256, c_ln_b + i * 256, xq, 1e-6f);
    gemmF(xq, c_qw + (size_t)i * 65536, nullptr, nullptr, qb, 4096, 256, 256, 0);
    gemmF(rgbf, c_kw + (size_t)i * 65536, nullptr, nullptr, bufK, 32768, 256, 256, 0);
    gemmF(rgbf, c_vw + (size_t)i * 65536, nullptr, nullptr, bufV, 32768, 256, 256, 0);
    gemmG(ray_diff, c_pw1 + (size_t)i * 128, c_pb1 + i * 32, nullptr, ahid, 32768, 4, 32, 1);
    gemmF(ahid, c_pw2 + (size_t)i * 8192, c_pb2 + i * 256, nullptr, bufP, 32768, 32, 256, 0);
    tdiff_kernel<<<4096, 256, 0, stream>>>(bufK, qb, bufP, bufT);
    gemmF(bufT, c_aw1 + (size_t)i * 8192, c_ab1 + i * 32, nullptr, ahid, 32768, 256, 32, 1);
    gemmF(ahid, c_aw2 + (size_t)i * 8192, c_ab2 + i * 256, nullptr, bufT, 32768, 32, 256, 0);
    cross_combine_kernel<<<512, 256, 0, stream>>>(bufT, bufV, bufP, mask, ob);
    gemmF(ob, c_ow + (size_t)i * 65536, c_ob + i * 256, fq, fq, 4096, 256, 256, 0);
    ln(fq, c_fg + i * 256, c_fb + i * 256, xq, 1e-6f);
    gemmF(xq, c_fw1 + (size_t)i * 262144, c_fb1 + i * 1024, nullptr, hid, 4096, 256, 1024, 1);
    gemmF(hid, c_fw2 + (size_t)i * 262144, c_fb2 + i * 256, fq, fq, 4096, 1024, 256, 0);

    // ---- q_fc on even layers ----
    if ((i & 1) == 0) {
      concat_kernel<<<(4096 * 346 + 255) / 256, 256, 0, stream>>>(fq, pe, ve, catb);
      gemmG(catb, q_w1 + (size_t)j * 346 * 256, q_b1 + j * 256, nullptr, qb, 4096, 346, 256, 1);
      gemmF(qb, q_w2 + (size_t)j * 65536, q_b2 + j * 256, nullptr, fq, 4096, 256, 256, 0);
      ++j;
    }

    // ---- self Transformer (MHA over samples) ----
    ln(fq, s_ln_g + i * 256, s_ln_b + i * 256, xq, 1e-6f);
    gemmF(xq, s_qw + (size_t)i * 65536, nullptr, nullptr, sq, 4096, 256, 256, 0);
    gemmF(xq, s_kw + (size_t)i * 65536, nullptr, nullptr, sk, 4096, 256, 256, 0);
    gemmF(xq, s_vw + (size_t)i * 65536, nullptr, nullptr, sv, 4096, 256, 256, 0);
    self_attn_kernel<<<256, 256, 0, stream>>>(sq, sk, sv, ob);
    gemmF(ob, s_ow + (size_t)i * 65536, s_ob + i * 256, fq, fq, 4096, 256, 256, 0);
    ln(fq, s_fg + i * 256, s_fb + i * 256, xq, 1e-6f);
    gemmF(xq, s_fw1 + (size_t)i * 262144, s_fb1 + i * 1024, nullptr, hid, 4096, 256, 1024, 1);
    gemmF(hid, s_fw2 + (size_t)i * 262144, s_fb2 + i * 256, fq, fq, 4096, 1024, 256, 0);
  }

  // ---- output head ----
  final_kernel<<<64, 256, 0, stream>>>(fq, norm_g, norm_b, out_w, out_b, (float*)d_out);
}
